// HGPSLEncoder_41369124995124
// MI455X (gfx1250) — matrix operand order repaired and checked by reading, not compiled, not run on hardware
//
#include <hip/hip_runtime.h>

// ---------------------------------------------------------------------------
// HGP-SL encoder forward for gfx1250 (MI455X, wave32, WMMA).
// Heavy batched einsums -> v_wmma_f32_16x16x32_bf16 (f32 accumulate).
//   A operand: 2x global_load_b128 (contiguous in K).
//   B operand: rows loaded as b128, staged COLUMN-MAJOR in wave-private LDS
//              (scalar ds_store_b16 scatter, no waits), consumed as
//              2x ds_load_b128 per lane (single s_wait_dscnt).
//   4 WMMAs share each B operand (64x16 strip per wave).
// Irregular parts (scatter, top-k, sparsemax, MLP) -> fp32 VALU.
// ---------------------------------------------------------------------------

typedef __attribute__((ext_vector_type(16))) __bf16 v16bf;
typedef __attribute__((ext_vector_type(8)))  __bf16 bf16x8;
typedef __attribute__((ext_vector_type(8)))  float  v8f;

constexpr int  B_   = 256;
constexpr int  N_   = 512;
constexpr int  FEAT_= 3;
constexpr int  H_   = 128;
constexpr int  EMB_ = 64;
constexpr int  DEG_ = 16;
constexpr int  K1_  = 256;
constexpr int  K2_  = 128;
constexpr long E_   = (long)B_ * N_ * DEG_;   // 2,097,152 edges
constexpr float LAMB_ = 1.0f;

// ------------------------------ utilities ---------------------------------

__global__ void k_fill_f32(float* p, size_t n, float v) {
  size_t i = (size_t)blockIdx.x * blockDim.x + threadIdx.x;
  if (i < n) p[i] = v;
}

__global__ void k_fill_bf16(__bf16* p, size_t n, float v) {
  size_t i = (size_t)blockIdx.x * blockDim.x + threadIdx.x;
  if (i < n) p[i] = (__bf16)v;
}

__global__ void k_f32_to_bf16(const float* __restrict__ s, __bf16* __restrict__ d, size_t n) {
  size_t i = (size_t)blockIdx.x * blockDim.x + threadIdx.x;
  if (i < n) d[i] = (__bf16)s[i];
}

// ----------------------- batched bf16 WMMA GEMM ---------------------------
// C[b](MxNd) = A[b](MxK) * Bm[b](KxNd), row-major, f32 accumulate.
// One wave computes a 64x16 strip (4 vertical 16x16 tiles): the B operand is
// fetched once per 4 WMMAs. B tile staged column-major in wave-private LDS:
//   store side: lane L holds row k0+L (2x global b128) -> 16x ds_store_b16
//   load  side: lane r reads col r, K=hi*16..+15 -> 2x ds_load_b128
// sB==0 broadcasts Bm. Launch 256 threads (8 waves). M%64==0, Nd%16==0, K%32==0.

__global__ void k_bgemm_bf16(const __bf16* __restrict__ A,
                             const __bf16* __restrict__ Bm,
                             float* __restrict__ C,
                             int M, int Nd, int K,
                             long sA, long sB, long sC)
{
  __shared__ __bf16 bst[8][16 * 32];              // per-wave: 16 cols x 32 ks (1KB)
  const int lane   = threadIdx.x & 31;
  const int wave   = threadIdx.x >> 5;
  const int tilesN = Nd >> 4;
  const int strips = M  >> 6;                     // 64-row strips
  const int task   = blockIdx.x * 8 + wave;
  if (task >= strips * tilesN) return;            // wave-uniform: EXEC stays full
  const int sm = task / tilesN;
  const int tn = task - sm * tilesN;
  const int b  = blockIdx.y;

  const __bf16* Ab = A  + (long)b * sA + (long)(sm * 64) * K;
  const __bf16* Bb = Bm + (long)b * sB + (long)(tn * 16);
  float*        Cb = C  + (long)b * sC + (long)(sm * 64) * Nd + (long)(tn * 16);

  const int r  = lane & 15;
  const int hi = lane >> 4;
  __bf16* myB = bst[wave];

  v8f zero = {};
  v8f acc[4] = { zero, zero, zero, zero };

  for (int k0 = 0; k0 < K; k0 += 32) {
    // ---- stage B tile (column-major): lane loads row (k0+lane) ----
    const __bf16* brow = Bb + (long)(k0 + lane) * Nd;
    __builtin_prefetch(brow + 32 * Nd, 0, 3);     // next B rows
    bf16x8 rv0 = *(const bf16x8*)(brow);          // cols 0..7
    bf16x8 rv1 = *(const bf16x8*)(brow + 8);      // cols 8..15
#pragma unroll
    for (int c = 0; c < 8; ++c) {
      myB[c * 32 + lane]       = rv0[c];          // ds_store_b16 scatter
      myB[(c + 8) * 32 + lane] = rv1[c];
    }
    // wave-private region; LDS ops from one wave are in-order (DScnt).

    // ---- B operand: lane holds column r, K = hi*16 + e (contiguous) ----
    bf16x8 c0 = *(const bf16x8*)(myB + r * 32 + hi * 16);      // ds_load_b128
    bf16x8 c1 = *(const bf16x8*)(myB + r * 32 + hi * 16 + 8);  // ds_load_b128
    v16bf bv;
#pragma unroll
    for (int e = 0; e < 8; ++e) { bv[e] = c0[e]; bv[e + 8] = c1[e]; }

    // ---- 4 M-tiles share this B operand ----
    const __bf16* ar = Ab + (long)r * K + k0;
#pragma unroll
    for (int t = 0; t < 4; ++t) {
      const __bf16* art = ar + (long)(t * 16) * K;
      __builtin_prefetch(art + 32, 0, 3);         // next A chunk
      bf16x8 a0 = *(const bf16x8*)(art + hi * 16);       // K = hi*16 + 0..7
      bf16x8 a1 = *(const bf16x8*)(art + hi * 16 + 8);   // K = hi*16 + 8..15
      v16bf av;
#pragma unroll
      for (int e = 0; e < 8; ++e) { av[e] = a0[e]; av[e + 8] = a1[e]; }
      acc[t] = __builtin_amdgcn_wmma_f32_16x16x32_bf16(
          false, av, false, bv, (short)0, acc[t], false, false);
    }
  }

#pragma unroll
  for (int t = 0; t < 4; ++t) {
    float* crow = Cb + (long)(t * 16) * Nd;
#pragma unroll
    for (int i = 0; i < 8; ++i)                   // D: row i + 8*hi, col r
      crow[(long)(i + 8 * hi) * Nd + r] = acc[t][i];
  }
}

// --------------------------- edge-list GCN --------------------------------

__global__ void k_xw(const float* __restrict__ x, const float* __restrict__ W1,
                     float* __restrict__ xw) {
  size_t gid = (size_t)blockIdx.x * blockDim.x + threadIdx.x;
  if (gid >= (size_t)B_ * N_ * H_) return;
  int c = gid % H_; size_t n = gid / H_;
  float a = 0.f;
#pragma unroll
  for (int f = 0; f < FEAT_; ++f) a += x[n * FEAT_ + f] * W1[f * H_ + c];
  xw[gid] = a;
}

__global__ void k_deg(const int* __restrict__ dst, float* deg, long E) {
  long e = (long)blockIdx.x * blockDim.x + threadIdx.x;
  if (e < E) atomicAdd(&deg[dst[e]], 1.0f);
}

__global__ void k_msg(const int* __restrict__ src, const int* __restrict__ dst,
                      const float* __restrict__ xw, const float* __restrict__ deg,
                      float* __restrict__ agg, long E) {
  size_t gid = (size_t)blockIdx.x * blockDim.x + threadIdx.x;
  if (gid >= (size_t)E * H_) return;
  long e = gid >> 7; int c = gid & (H_ - 1);
  int s = src[e], d = dst[e];
  float w = rsqrtf(deg[s]) * rsqrtf(deg[d]);
  atomicAdd(&agg[(size_t)d * H_ + c], xw[(size_t)s * H_ + c] * w);
}

__global__ void k_gcn_combine(const float* __restrict__ agg, const float* __restrict__ xw,
                              const float* __restrict__ deg, const float* __restrict__ b1,
                              float* __restrict__ h, __bf16* __restrict__ hbf) {
  size_t gid = (size_t)blockIdx.x * blockDim.x + threadIdx.x;
  if (gid >= (size_t)B_ * N_ * H_) return;
  int c = gid % H_; size_t n = gid / H_;
  float v = agg[gid] + xw[gid] / deg[n] + b1[c];
  v = fmaxf(v, 0.f);
  h[gid] = v; hbf[gid] = (__bf16)v;
}

__global__ void k_set_adj(const int* __restrict__ src, const int* __restrict__ dst,
                          __bf16* __restrict__ A, long E) {
  long e = (long)blockIdx.x * blockDim.x + threadIdx.x;
  if (e >= E) return;
  int s = src[e], d = dst[e];
  int g = s / N_;
  A[(size_t)g * N_ * N_ + (size_t)(s % N_) * N_ + (d % N_)] = (__bf16)1.0f;
}

// --------------------------- pooling helpers ------------------------------

__global__ void k_rowsum_bf16(const __bf16* __restrict__ A, float* rs, int n) {
  size_t gid = (size_t)blockIdx.x * blockDim.x + threadIdx.x;
  if (gid >= (size_t)B_ * n) return;
  size_t b = gid / n, i = gid % n;
  const __bf16* row = A + ((size_t)b * n + i) * n;
  float s = 0.f;
  for (int j = 0; j < n; ++j) s += (float)row[j];
  rs[gid] = s;
}

__global__ void k_rowsum_f32(const float* __restrict__ A, float* rs, int n) {
  size_t gid = (size_t)blockIdx.x * blockDim.x + threadIdx.x;
  if (gid >= (size_t)B_ * n) return;
  size_t b = gid / n, i = gid % n;
  const float* row = A + ((size_t)b * n + i) * n;
  float s = 0.f;
  for (int j = 0; j < n; ++j) s += row[j];
  rs[gid] = s;
}

// score[b,i] = sum_c | x[b,i,c] - neigh_raw[b,i,c]/max(rowsum,1) |
__global__ void k_score(const float* __restrict__ xd, const float* __restrict__ neigh,
                        const float* __restrict__ rs, float* score, int n) {
  size_t gid = (size_t)blockIdx.x * blockDim.x + threadIdx.x;
  if (gid >= (size_t)B_ * n) return;
  float dg = fmaxf(rs[gid], 1.0f);
  const float* xr = xd    + gid * H_;
  const float* nr = neigh + gid * H_;
  float s = 0.f;
  for (int c = 0; c < H_; ++c) s += fabsf(xr[c] - nr[c] / dg);
  score[gid] = s;
}

// Per-graph top-k via in-LDS bitonic sort (descending). blockDim == NS.
template <int NS>
__global__ void k_topk(const float* __restrict__ score, int* __restrict__ idx_out, int k) {
  __shared__ float sv[NS];
  __shared__ int   sid[NS];
  int tid = threadIdx.x, b = blockIdx.x;
  sv[tid] = score[(size_t)b * NS + tid];
  sid[tid] = tid;
  __syncthreads();
  for (int ksz = 2; ksz <= NS; ksz <<= 1)
    for (int j = ksz >> 1; j > 0; j >>= 1) {
      int ixj = tid ^ j;
      if (ixj > tid) {
        bool desc = ((tid & ksz) == 0);
        float a0 = sv[tid], a1 = sv[ixj];
        bool sw = desc ? (a0 < a1) : (a0 > a1);
        if (sw) { sv[tid] = a1; sv[ixj] = a0;
                  int t = sid[tid]; sid[tid] = sid[ixj]; sid[ixj] = t; }
      }
      __syncthreads();
    }
  if (tid < k) idx_out[(size_t)b * k + tid] = sid[tid];
}

__global__ void k_gather_rows(const float* __restrict__ xd, const int* __restrict__ idx,
                              float* __restrict__ xk, __bf16* __restrict__ xkbf,
                              int n, int k) {
  size_t gid = (size_t)blockIdx.x * blockDim.x + threadIdx.x;
  if (gid >= (size_t)B_ * k * H_) return;
  int c = gid % H_; size_t rem = gid / H_;
  int i = rem % k; int b = rem / k;
  int node = idx[(size_t)b * k + i];
  float v = xd[((size_t)b * n + node) * H_ + c];
  xk[gid] = v; xkbf[gid] = (__bf16)v;
}

__global__ void k_att(const float* __restrict__ xk, const float* __restrict__ att,
                      float* __restrict__ si, float* __restrict__ sj, int k) {
  size_t gid = (size_t)blockIdx.x * blockDim.x + threadIdx.x;
  if (gid >= (size_t)B_ * k) return;
  const float* xr = xk + gid * H_;
  float a = 0.f, b = 0.f;
  for (int c = 0; c < H_; ++c) { a += xr[c] * att[c]; b += xr[c] * att[H_ + c]; }
  si[gid] = a; sj[gid] = b;
}

// One block per row: e_j = lrelu(si+sj,0.2) + LAMB*adj_k ; sparsemax over j.
// srcB!=null -> gather bf16 source adjacency, else f32. blockDim == NK.
template <int NK>
__global__ void k_sparsemax(const float* __restrict__ si, const float* __restrict__ sj,
                            const float* __restrict__ srcF, const __bf16* __restrict__ srcB,
                            int n_src, const int* __restrict__ idx,
                            float* __restrict__ out, float lamb) {
  __shared__ float zs[NK];
  __shared__ float cs[NK];
  __shared__ int   cnt;
  int tid = threadIdx.x;
  int row = blockIdx.x;                 // b*NK + i
  int b = row / NK, i = row - b * NK;
  if (tid == 0) cnt = 0;
  int ni = idx[(size_t)b * NK + i];
  int nj = idx[(size_t)b * NK + tid];
  size_t aoff = (size_t)b * n_src * n_src + (size_t)ni * n_src + nj;
  float adjv = srcB ? (float)srcB[aoff] : srcF[aoff];
  float sv = si[(size_t)b * NK + i] + sj[(size_t)b * NK + tid];
  float z  = (sv >= 0.f ? sv : 0.2f * sv) + lamb * adjv;   // leaky_relu + LAMB*adj
  zs[tid] = z;
  __syncthreads();
  // bitonic sort, descending
  for (int ksz = 2; ksz <= NK; ksz <<= 1)
    for (int j = ksz >> 1; j > 0; j >>= 1) {
      int ixj = tid ^ j;
      if (ixj > tid) {
        bool desc = ((tid & ksz) == 0);
        float a0 = zs[tid], a1 = zs[ixj];
        bool sw = desc ? (a0 < a1) : (a0 > a1);
        if (sw) { zs[tid] = a1; zs[ixj] = a0; }
      }
      __syncthreads();
    }
  float m   = zs[0];
  float zsh = zs[tid] - m;
  cs[tid] = zsh;
  __syncthreads();
  for (int off = 1; off < NK; off <<= 1) {      // inclusive scan
    float t = (tid >= off) ? cs[tid - off] : 0.f;
    __syncthreads();
    cs[tid] += t;
    __syncthreads();
  }
  if (1.0f + (float)(tid + 1) * zsh > cs[tid]) atomicAdd(&cnt, 1);
  __syncthreads();
  int kmax = cnt;
  float tau = (cs[kmax - 1] - 1.0f) / (float)kmax;
  out[(size_t)row * NK + tid] = fmaxf(z - m - tau, 0.f);
}

// An = (adj + I) * rsqrt(d_i) * rsqrt(d_j), d = max(rowsum+1, 1e-12)  (bf16)
__global__ void k_make_An(const float* __restrict__ adj, const float* __restrict__ rs,
                          __bf16* __restrict__ An, int n) {
  size_t gid = (size_t)blockIdx.x * blockDim.x + threadIdx.x;
  if (gid >= (size_t)B_ * n * n) return;
  int j = gid % n; size_t rem = gid / n;
  int i = rem % n; int b = rem / n;
  float di = rsqrtf(fmaxf(rs[(size_t)b * n + i] + 1.0f, 1e-12f));
  float dj = rsqrtf(fmaxf(rs[(size_t)b * n + j] + 1.0f, 1e-12f));
  float a  = adj[gid] + (i == j ? 1.0f : 0.0f);
  An[gid] = (__bf16)(a * di * dj);
}

__global__ void k_bias_relu(const float* __restrict__ in, const float* __restrict__ bias,
                            float* __restrict__ out, __bf16* __restrict__ obf, size_t total) {
  size_t gid = (size_t)blockIdx.x * blockDim.x + threadIdx.x;
  if (gid >= total) return;
  int c = gid % H_;
  float v = fmaxf(in[gid] + bias[c], 0.f);
  out[gid] = v;
  if (obf) obf[gid] = (__bf16)v;
}

// readout: [B,k,H] -> [B,2H] = concat(max over k, mean over k)
__global__ void k_readout(const float* __restrict__ xk, float* __restrict__ out, int k) {
  size_t gid = (size_t)blockIdx.x * blockDim.x + threadIdx.x;
  if (gid >= (size_t)B_ * H_) return;
  int c = gid % H_, b = gid / H_;
  const float* base = xk + (size_t)b * k * H_ + c;
  float mx = -3.4e38f, s = 0.f;
  for (int i = 0; i < k; ++i) { float v = base[(size_t)i * H_]; mx = fmaxf(mx, v); s += v; }
  out[(size_t)b * 2 * H_ + c]      = mx;
  out[(size_t)b * 2 * H_ + H_ + c] = s / (float)k;
}

__global__ void k_zsum(const float* x1, const float* x2, const float* x3, float* z) {
  size_t gid = (size_t)blockIdx.x * blockDim.x + threadIdx.x;
  if (gid >= (size_t)B_ * 2 * H_) return;
  z[gid] = fmaxf(x1[gid], 0.f) + fmaxf(x2[gid], 0.f) + fmaxf(x3[gid], 0.f);
}

__global__ void k_linear(const float* __restrict__ in, const float* __restrict__ W,
                         const float* __restrict__ bias, float* __restrict__ out,
                         int In, int Out, int doRelu) {
  size_t gid = (size_t)blockIdx.x * blockDim.x + threadIdx.x;
  if (gid >= (size_t)B_ * Out) return;
  int o = gid % Out, b = gid / Out;
  const float* ir = in + (size_t)b * In;
  float a = bias[o];
  for (int i = 0; i < In; ++i) a += ir[i] * W[(size_t)i * Out + o];
  out[gid] = doRelu ? fmaxf(a, 0.f) : a;
}

__global__ void k_norm(const float* __restrict__ z, float* __restrict__ out) {
  int b = blockIdx.x * blockDim.x + threadIdx.x;
  if (b >= B_) return;
  float s = 0.f;
  for (int c = 0; c < EMB_; ++c) { float v = z[(size_t)b * EMB_ + c]; s += v * v; }
  float nrm = fmaxf(sqrtf(s), 1e-12f);
  for (int c = 0; c < EMB_; ++c) out[(size_t)b * EMB_ + c] = z[(size_t)b * EMB_ + c] / nrm;
}

// ------------------------------- host side --------------------------------

static inline unsigned nb(size_t n, int t) { return (unsigned)((n + t - 1) / t); }

extern "C" void kernel_launch(void* const* d_in, const int* in_sizes, int n_in,
                              void* d_out, int out_size, void* d_ws, size_t ws_size,
                              hipStream_t stream) {
  const float* x     = (const float*)d_in[0];
  const int*   ei    = (const int*)  d_in[1];
  const float* W1    = (const float*)d_in[2];
  const float* b1    = (const float*)d_in[3];
  const float* W2    = (const float*)d_in[4];
  const float* b2    = (const float*)d_in[5];
  const float* W3    = (const float*)d_in[6];
  const float* b3    = (const float*)d_in[7];
  const float* att1  = (const float*)d_in[8];
  const float* att2  = (const float*)d_in[9];
  const float* l1w   = (const float*)d_in[10];
  const float* l1b   = (const float*)d_in[11];
  const float* l2w   = (const float*)d_in[12];
  const float* l2b   = (const float*)d_in[13];
  const float* l3w   = (const float*)d_in[14];
  const float* l3b   = (const float*)d_in[15];
  const int* src = ei;
  const int* dst = ei + E_;

  char* w = (char*)d_ws; size_t off = 0;
  auto alloc = [&](size_t bytes) -> void* {
    void* p = w + off; off += (bytes + 255) & ~(size_t)255; return p;
  };

  float*  xw     = (float*) alloc((size_t)B_*N_*H_*4);
  float*  agg    = (float*) alloc((size_t)B_*N_*H_*4);   // reused as neigh1
  float*  h      = (float*) alloc((size_t)B_*N_*H_*4);
  __bf16* hbf    = (__bf16*)alloc((size_t)B_*N_*H_*2);
  __bf16* Abf    = (__bf16*)alloc((size_t)B_*N_*N_*2);
  float*  deg    = (float*) alloc((size_t)B_*N_*4);
  float*  rs1    = (float*) alloc((size_t)B_*N_*4);
  float*  score1 = (float*) alloc((size_t)B_*N_*4);
  int*    idx1   = (int*)   alloc((size_t)B_*K1_*4);
  float*  x1p    = (float*) alloc((size_t)B_*K1_*H_*4);
  __bf16* x1pbf  = (__bf16*)alloc((size_t)B_*K1_*H_*2);
  float*  si1    = (float*) alloc((size_t)B_*K1_*4);
  float*  sj1    = (float*) alloc((size_t)B_*K1_*4);
  float*  adj1   = (float*) alloc((size_t)B_*K1_*K1_*4);
  __bf16* adj1bf = (__bf16*)alloc((size_t)B_*K1_*K1_*2);
  float*  ssum1  = (float*) alloc((size_t)B_*K1_*4);
  __bf16* An1bf  = (__bf16*)alloc((size_t)B_*K1_*K1_*2);
  __bf16* W2bf   = (__bf16*)alloc((size_t)H_*H_*2);
  __bf16* W3bf   = (__bf16*)alloc((size_t)H_*H_*2);
  float*  xw2    = (float*) alloc((size_t)B_*K1_*H_*4);  // also reused as h2pre
  __bf16* xw2bf  = (__bf16*)alloc((size_t)B_*K1_*H_*2);
  float*  h2     = (float*) alloc((size_t)B_*K1_*H_*4);
  __bf16* h2bf   = (__bf16*)alloc((size_t)B_*K1_*H_*2);
  float*  neigh2 = (float*) alloc((size_t)B_*K1_*H_*4);
  float*  score2 = (float*) alloc((size_t)B_*K1_*4);
  int*    idx2   = (int*)   alloc((size_t)B_*K2_*4);
  float*  x2p    = (float*) alloc((size_t)B_*K2_*H_*4);
  __bf16* x2pbf  = (__bf16*)alloc((size_t)B_*K2_*H_*2);
  float*  si2    = (float*) alloc((size_t)B_*K2_*4);
  float*  sj2    = (float*) alloc((size_t)B_*K2_*4);
  float*  adj2   = (float*) alloc((size_t)B_*K2_*K2_*4);
  float*  ssum2  = (float*) alloc((size_t)B_*K2_*4);
  __bf16* An2bf  = (__bf16*)alloc((size_t)B_*K2_*K2_*2);
  float*  xw3    = (float*) alloc((size_t)B_*K2_*H_*4);  // also reused as h3pre
  __bf16* xw3bf  = (__bf16*)alloc((size_t)B_*K2_*H_*2);
  float*  h3     = (float*) alloc((size_t)B_*K2_*H_*4);
  float*  x1r    = (float*) alloc((size_t)B_*2*H_*4);
  float*  x2r    = (float*) alloc((size_t)B_*2*H_*4);
  float*  x3r    = (float*) alloc((size_t)B_*2*H_*4);
  float*  zb     = (float*) alloc((size_t)B_*2*H_*4);
  float*  z1     = (float*) alloc((size_t)B_*H_*4);
  float*  z2     = (float*) alloc((size_t)B_*H_*4);
  float*  z3     = (float*) alloc((size_t)B_*EMB_*4);
  (void)ws_size; (void)n_in; (void)in_sizes; (void)out_size;

  auto gemm = [&](const __bf16* A, const __bf16* Bm, float* C,
                  int M, int Nd, int K, long sA, long sB, long sC) {
    int tasks = (M / 64) * (Nd / 16);             // 64x16 strip per wave
    dim3 grid((unsigned)((tasks + 7) / 8), (unsigned)B_);
    k_bgemm_bf16<<<grid, 256, 0, stream>>>(A, Bm, C, M, Nd, K, sA, sB, sC);
  };

  const size_t NH = (size_t)B_ * N_ * H_;

  // ---- edge-list GCN: h = relu(gcn_edge(x, W1, b1)) ----
  k_xw<<<nb(NH, 256), 256, 0, stream>>>(x, W1, xw);
  k_fill_f32<<<nb((size_t)B_*N_, 256), 256, 0, stream>>>(deg, (size_t)B_*N_, 1.0f);
  k_deg<<<nb((size_t)E_, 256), 256, 0, stream>>>(dst, deg, E_);
  k_fill_f32<<<nb(NH, 256), 256, 0, stream>>>(agg, NH, 0.0f);
  k_msg<<<nb((size_t)E_*H_, 256), 256, 0, stream>>>(src, dst, xw, deg, agg, E_);
  k_gcn_combine<<<nb(NH, 256), 256, 0, stream>>>(agg, xw, deg, b1, h, hbf);

  // ---- dense adjacency (bf16) ----
  k_fill_bf16<<<nb((size_t)B_*N_*N_, 256), 256, 0, stream>>>(Abf, (size_t)B_*N_*N_, 0.0f);
  k_set_adj<<<nb((size_t)E_, 256), 256, 0, stream>>>(src, dst, Abf, E_);
  k_rowsum_bf16<<<nb((size_t)B_*N_, 256), 256, 0, stream>>>(Abf, rs1, N_);

  // ---- pool 1: neigh = A @ h (WMMA), score, top-K1, gather, sparsemax ----
  gemm(Abf, hbf, agg, N_, H_, N_, (long)N_*N_, (long)N_*H_, (long)N_*H_);  // neigh1 in agg
  k_score<<<nb((size_t)B_*N_, 256), 256, 0, stream>>>(h, agg, rs1, score1, N_);
  k_topk<N_><<<B_, N_, 0, stream>>>(score1, idx1, K1_);
  k_gather_rows<<<nb((size_t)B_*K1_*H_, 256), 256, 0, stream>>>(h, idx1, x1p, x1pbf, N_, K1_);
  k_att<<<nb((size_t)B_*K1_, 256), 256, 0, stream>>>(x1p, att1, si1, sj1, K1_);
  k_sparsemax<K1_><<<B_*K1_, K1_, 0, stream>>>(si1, sj1, nullptr, Abf, N_, idx1, adj1, LAMB_);
  k_readout<<<nb((size_t)B_*H_, 256), 256, 0, stream>>>(x1p, x1r, K1_);

  // ---- gcn_dense 1: h2 = relu(An1 @ (x1p @ W2) + b2) (two WMMA GEMMs) ----
  k_rowsum_f32<<<nb((size_t)B_*K1_, 256), 256, 0, stream>>>(adj1, ssum1, K1_);
  k_make_An<<<nb((size_t)B_*K1_*K1_, 256), 256, 0, stream>>>(adj1, ssum1, An1bf, K1_);
  k_f32_to_bf16<<<nb((size_t)H_*H_, 256), 256, 0, stream>>>(W2, W2bf, (size_t)H_*H_);
  gemm(x1pbf, W2bf, xw2, K1_, H_, H_, (long)K1_*H_, 0, (long)K1_*H_);
  k_f32_to_bf16<<<nb((size_t)B_*K1_*H_, 256), 256, 0, stream>>>(xw2, xw2bf, (size_t)B_*K1_*H_);
  gemm(An1bf, xw2bf, xw2, K1_, H_, K1_, (long)K1_*K1_, (long)K1_*H_, (long)K1_*H_); // h2pre
  k_bias_relu<<<nb((size_t)B_*K1_*H_, 256), 256, 0, stream>>>(xw2, b2, h2, h2bf, (size_t)B_*K1_*H_);

  // ---- pool 2: neigh2 = adj1 @ h2 (WMMA), score, top-K2, gather, sparsemax ----
  k_f32_to_bf16<<<nb((size_t)B_*K1_*K1_, 256), 256, 0, stream>>>(adj1, adj1bf, (size_t)B_*K1_*K1_);
  gemm(adj1bf, h2bf, neigh2, K1_, H_, K1_, (long)K1_*K1_, (long)K1_*H_, (long)K1_*H_);
  k_score<<<nb((size_t)B_*K1_, 256), 256, 0, stream>>>(h2, neigh2, ssum1, score2, K1_);
  k_topk<K1_><<<B_, K1_, 0, stream>>>(score2, idx2, K2_);
  k_gather_rows<<<nb((size_t)B_*K2_*H_, 256), 256, 0, stream>>>(h2, idx2, x2p, x2pbf, K1_, K2_);
  k_att<<<nb((size_t)B_*K2_, 256), 256, 0, stream>>>(x2p, att2, si2, sj2, K2_);
  k_sparsemax<K2_><<<B_*K2_, K2_, 0, stream>>>(si2, sj2, adj1, nullptr, K1_, idx2, adj2, LAMB_);
  k_readout<<<nb((size_t)B_*H_, 256), 256, 0, stream>>>(x2p, x2r, K2_);

  // ---- gcn_dense 2: h3 = relu(An2 @ (x2p @ W3) + b3) ----
  k_rowsum_f32<<<nb((size_t)B_*K2_, 256), 256, 0, stream>>>(adj2, ssum2, K2_);
  k_make_An<<<nb((size_t)B_*K2_*K2_, 256), 256, 0, stream>>>(adj2, ssum2, An2bf, K2_);
  k_f32_to_bf16<<<nb((size_t)H_*H_, 256), 256, 0, stream>>>(W3, W3bf, (size_t)H_*H_);
  gemm(x2pbf, W3bf, xw3, K2_, H_, H_, (long)K2_*H_, 0, (long)K2_*H_);
  k_f32_to_bf16<<<nb((size_t)B_*K2_*H_, 256), 256, 0, stream>>>(xw3, xw3bf, (size_t)B_*K2_*H_);
  gemm(An2bf, xw3bf, xw3, K2_, K2_ == K2_ ? H_ : H_, K2_, (long)K2_*K2_, (long)K2_*H_, (long)K2_*H_); // h3pre
  k_bias_relu<<<nb((size_t)B_*K2_*H_, 256), 256, 0, stream>>>(xw3, b3, h3, nullptr, (size_t)B_*K2_*H_);
  k_readout<<<nb((size_t)B_*H_, 256), 256, 0, stream>>>(h3, x3r, K2_);

  // ---- head: z = relu(x1)+relu(x2)+relu(x3); MLP; L2 normalize ----
  k_zsum<<<nb((size_t)B_*2*H_, 256), 256, 0, stream>>>(x1r, x2r, x3r, zb);
  k_linear<<<nb((size_t)B_*H_, 256), 256, 0, stream>>>(zb, l1w, l1b, z1, 2*H_, H_, 1);
  k_linear<<<nb((size_t)B_*H_, 256), 256, 0, stream>>>(z1, l2w, l2b, z2, H_, H_, 1);
  k_linear<<<nb((size_t)B_*EMB_, 256), 256, 0, stream>>>(z2, l3w, l3b, z3, H_, EMB_, 0);
  k_norm<<<1, B_, 0, stream>>>(z3, (float*)d_out);
}
